// RAMCrossAttention_22832046146305
// MI455X (gfx1250) — 8-bit matrix operand order repaired and checked by reading, compiled, not run on hardware
//
#include <hip/hip_runtime.h>
#include <hip/hip_bf16.h>
#include <cstdint>

#define NQ   4096
#define NK   4096
#define HH   4
#define DB   64
#define EB   64
#define NPOS 13

typedef __attribute__((ext_vector_type(16))) int   v16i;
typedef __attribute__((ext_vector_type(8)))  float v8f;

#if defined(__HIP_DEVICE_COMPILE__)
#if __has_builtin(__builtin_amdgcn_global_load_async_to_lds_b128) && \
    __has_builtin(__builtin_amdgcn_s_wait_asynccnt)
#define USE_ASYNC_LDS 1
#endif
#endif
#ifndef USE_ASYNC_LDS
#define USE_ASYNC_LDS 0
#endif

#if USE_ASYNC_LDS
typedef int v4i __attribute__((__vector_size__(4 * sizeof(int))));
typedef __attribute__((address_space(1))) v4i v4i_g;
typedef __attribute__((address_space(3))) v4i v4i_l;
#endif

__device__ __forceinline__ v8f v8fzero() {
    v8f v;
#pragma unroll
    for (int i = 0; i < 8; ++i) v[i] = 0.0f;
    return v;
}

__device__ __forceinline__ int spread_fp8(unsigned x) {
    unsigned v = x & 0xFu;
    unsigned m = (v | (v << 7) | (v << 14) | (v << 21)) & 0x01010101u;
    return (int)(m * 0x38u);
}

__global__ void simbits_kernel(const float* __restrict__ sim_table,
                               unsigned* __restrict__ sim_bits) {
    int tid = blockIdx.x * blockDim.x + threadIdx.x;
    if (tid >= HH * 128) return;
    int h = tid >> 7, w = tid & 127;
    const float* p = sim_table + h * 4096 + w * 32;
    unsigned word = 0;
#pragma unroll
    for (int i = 0; i < 32; ++i) word |= (p[i] > 0.5f ? 1u : 0u) << i;
    sim_bits[tid] = word;
}

__global__ void addrq_kernel(const int* __restrict__ dec_bits,
                             const int* __restrict__ sim_conn,
                             int* __restrict__ addr_q) {
    int idx = blockIdx.x * blockDim.x + threadIdx.x;
    if (idx >= HH * NQ) return;
    int h = idx / NQ, q = idx % NQ;
    int addr = 0;
#pragma unroll
    for (int j = 0; j < 12; ++j) {
        int c = sim_conn[h * 12 + j];
        if (c < DB) addr += dec_bits[q * DB + c] << j;
    }
    addr_q[idx] = addr;
}

__global__ void addrkp_kernel(const int* __restrict__ enc_bits,
                              const int* __restrict__ sim_conn,
                              int* __restrict__ addr_kp) {
    int idx = blockIdx.x * blockDim.x + threadIdx.x;
    if (idx >= HH * NK) return;
    int h = idx / NK, k = idx % NK;
    int addr = 0;
#pragma unroll
    for (int j = 0; j < 12; ++j) {
        int c = sim_conn[h * 12 + j];
        if (c >= DB) {
            int cc = c - DB;
            int bit = (cc < EB) ? enc_bits[k * EB + cc]
                                : ((k >> (NPOS - 1 - (cc - EB))) & 1);
            addr += bit << j;
        }
    }
    addr_kp[idx] = addr;
}

__global__ void __launch_bounds__(256)
vals_kernel(const int* __restrict__ enc_bits, const int* __restrict__ val_conn,
            const float* __restrict__ val_table, unsigned char* __restrict__ valsT) {
    int k = blockIdx.x * 256 + threadIdx.x;
    int d = blockIdx.y, h = blockIdx.z;
    const int* vc = val_conn + (h * DB + d) * 10;
    int addr = 0;
#pragma unroll
    for (int j = 0; j < 10; ++j) {
        int c = vc[j];
        int bit = (c < EB) ? enc_bits[k * EB + c]
                           : ((k >> (NPOS - 1 - (c - EB))) & 1);
        addr += bit << j;
    }
    float v = val_table[(((size_t)(h * DB + d)) << 10) + addr];
    valsT[(((size_t)(h * DB + d)) << 12) + k] = (v > 0.5f) ? 0x38 : 0x00;
}

__global__ void __launch_bounds__(128)
attbits_kernel(const int* __restrict__ addr_q, const int* __restrict__ addr_kp,
               const unsigned* __restrict__ sim_bits, unsigned* __restrict__ att_bits) {
    __shared__ int      s_akp[NK];
    __shared__ unsigned s_sb[128];
    const int h   = blockIdx.y;
    const int tid = threadIdx.x;
    for (int i = tid; i < NK; i += 128) s_akp[i] = addr_kp[h * NK + i];
    if (tid < 128) s_sb[tid] = sim_bits[h * 128 + tid];
    __syncthreads();
    for (int qi = 0; qi < 32; ++qi) {
        int q  = blockIdx.x * 32 + qi;
        int aq = addr_q[h * NQ + q];
        unsigned word = 0;
#pragma unroll 8
        for (int i = 0; i < 32; ++i) {
            int a = aq + s_akp[tid * 32 + i];
            word |= ((s_sb[a >> 5] >> (a & 31)) & 1u) << i;
        }
        att_bits[((size_t)(h * NQ + q)) * 128 + tid] = word;
    }
}

__device__ __forceinline__ void stage_panel(const unsigned char* __restrict__ vbase,
                                            int kbase, unsigned char* ldsdst, int tid) {
#pragma unroll
    for (int i = 0; i < 4; ++i) {
        int c   = tid + 128 * i;
        int d   = c >> 3;
        int off = (c & 7) * 16;
        const unsigned char* g = vbase + (size_t)d * NK + kbase + off;
        unsigned char*       l = ldsdst + d * 128 + off;
#if USE_ASYNC_LDS
        __builtin_amdgcn_global_load_async_to_lds_b128(
            (v4i_g*)(uintptr_t)g, (v4i_l*)(unsigned)(uintptr_t)l, 0, 0);
#else
        *(uint4*)l = *(const uint4*)g;
#endif
    }
}

__global__ void __launch_bounds__(128)
gemm_kernel(const unsigned* __restrict__ att_bits,
            const unsigned char* __restrict__ valsT,
            unsigned char* __restrict__ combined) {
    __shared__ unsigned char sB[2][DB * 128];

    const int h     = blockIdx.z;
    const int tid   = threadIdx.x;
    const int lane  = tid & 31;
    const int wave  = tid >> 5;
    const int q0    = blockIdx.x * 64 + wave * 16;
    const int col   = lane & 15;
    const int khalf = lane >> 4;

    const unsigned* attb = att_bits + ((size_t)(h * NQ + q0 + col)) * 128;
    const unsigned char* vbase = valsT + ((size_t)h * DB) * NK;

    v8f acc[4];
#pragma unroll
    for (int j = 0; j < 4; ++j) acc[j] = v8fzero();
    int cntacc = 0;

    const unsigned s = (unsigned)(khalf * 8);

    stage_panel(vbase, 0, sB[0], tid);

    for (int t = 0; t < 32; ++t) {
        if (t + 1 < 32) {
            stage_panel(vbase, (t + 1) * 128, sB[(t + 1) & 1], tid);
#if USE_ASYNC_LDS
            __builtin_amdgcn_s_wait_asynccnt(4);
#endif
        } else {
#if USE_ASYNC_LDS
            __builtin_amdgcn_s_wait_asynccnt(0);
#endif
        }
        __syncthreads();
        const unsigned char* sbuf = sB[t & 1];

        uint4 ww = ((const uint4*)attb)[t];
        cntacc += __popc(ww.x) + __popc(ww.y) + __popc(ww.z) + __popc(ww.w);
        v16i a;
        a[0]  = spread_fp8(ww.x >> s);        a[1]  = spread_fp8(ww.x >> (s + 4));
        a[2]  = spread_fp8(ww.x >> (s + 16)); a[3]  = spread_fp8(ww.x >> (s + 20));
        a[4]  = spread_fp8(ww.y >> s);        a[5]  = spread_fp8(ww.y >> (s + 4));
        a[6]  = spread_fp8(ww.y >> (s + 16)); a[7]  = spread_fp8(ww.y >> (s + 20));
        a[8]  = spread_fp8(ww.z >> s);        a[9]  = spread_fp8(ww.z >> (s + 4));
        a[10] = spread_fp8(ww.z >> (s + 16)); a[11] = spread_fp8(ww.z >> (s + 20));
        a[12] = spread_fp8(ww.w >> s);        a[13] = spread_fp8(ww.w >> (s + 4));
        a[14] = spread_fp8(ww.w >> (s + 16)); a[15] = spread_fp8(ww.w >> (s + 20));

#pragma unroll
        for (int j = 0; j < 4; ++j) {
            const unsigned char* bp = sbuf + (j * 16 + col) * 128 + khalf * 8;
            v16i b;
#pragma unroll
            for (int seg = 0; seg < 4; ++seg) {
                uint2 u0 = *(const uint2*)(bp + seg * 32);
                uint2 u1 = *(const uint2*)(bp + seg * 32 + 16);
                b[seg * 4 + 0] = (int)u0.x; b[seg * 4 + 1] = (int)u0.y;
                b[seg * 4 + 2] = (int)u1.x; b[seg * 4 + 3] = (int)u1.y;
            }
            acc[j] = __builtin_amdgcn_wmma_f32_16x16x128_fp8_fp8(
                a, b, (short)0, acc[j], false, false);
        }
        __syncthreads();
    }

#pragma unroll
    for (int r = 0; r < 8; ++r) {
        int   cnt  = __shfl(cntacc, r + 8 * khalf, 32);
        float fcnt = (float)cnt;
        int   rowq = q0 + r + 8 * khalf;
#pragma unroll
        for (int j = 0; j < 4; ++j) {
            float sv = acc[j][r];
            unsigned char g = (cnt > 0 && 2.0f * sv >= fcnt) ? 1 : 0;
            combined[(size_t)rowq * (HH * DB) + h * DB + j * 16 + col] = g;
        }
    }
}

__global__ void out_kernel(const unsigned char* __restrict__ combined,
                           const int* __restrict__ out_conn,
                           const float* __restrict__ out_table,
                           float* __restrict__ out) {
    int idx = blockIdx.x * blockDim.x + threadIdx.x;
    if (idx >= NQ * DB) return;
    int q = idx >> 6, d = idx & 63;
    const int* oc = out_conn + d * 12;
    int addr = 0;
#pragma unroll
    for (int j = 0; j < 12; ++j)
        addr += (int)combined[(size_t)q * (HH * DB) + oc[j]] << j;
    out[idx] = (out_table[(d << 12) + addr] > 0.5f) ? 1.0f : 0.0f;
}

extern "C" void kernel_launch(void* const* d_in, const int* in_sizes, int n_in,
                              void* d_out, int out_size, void* d_ws, size_t ws_size,
                              hipStream_t stream) {
    const int*   dec_bits  = (const int*)d_in[0];
    const int*   enc_bits  = (const int*)d_in[1];
    const int*   sim_conn  = (const int*)d_in[2];
    const float* sim_table = (const float*)d_in[3];
    const int*   val_conn  = (const int*)d_in[4];
    const float* val_table = (const float*)d_in[5];
    const int*   out_conn  = (const int*)d_in[6];
    const float* out_table = (const float*)d_in[7];
    float*       out       = (float*)d_out;

    char* ws = (char*)d_ws;
    int*           addr_q   = (int*)(ws + 0);
    int*           addr_kp  = (int*)(ws + (64 << 10));
    unsigned*      sim_bits = (unsigned*)(ws + (128 << 10));
    unsigned char* valsT    = (unsigned char*)(ws + (256 << 10));
    unsigned*      att_bits = (unsigned*)(ws + (256 << 10) + (1 << 20));
    unsigned char* combined = (unsigned char*)(ws + (256 << 10) + (9 << 20));

    simbits_kernel<<<1, 512, 0, stream>>>(sim_table, sim_bits);
    addrq_kernel  <<<64, 256, 0, stream>>>(dec_bits, sim_conn, addr_q);
    addrkp_kernel <<<64, 256, 0, stream>>>(enc_bits, sim_conn, addr_kp);
    vals_kernel   <<<dim3(16, 64, 4), 256, 0, stream>>>(enc_bits, val_conn, val_table, valsT);
    attbits_kernel<<<dim3(128, 4), 128, 0, stream>>>(addr_q, addr_kp, sim_bits, att_bits);
    gemm_kernel   <<<dim3(64, 1, 4), 128, 0, stream>>>(att_bits, valsT, combined);
    out_kernel    <<<dim3(1024), 256, 0, stream>>>(combined, out_conn, out_table, out);
}
